// SCAM_69295002353919
// MI455X (gfx1250) — matrix operand order repaired and checked by reading, not compiled, not run on hardware
//
#include <hip/hip_runtime.h>
#include <hip/hip_bf16.h>

typedef __attribute__((ext_vector_type(16))) _Float16 v16h;
typedef __attribute__((ext_vector_type(8)))  _Float16 v8h;
typedef __attribute__((ext_vector_type(4)))  _Float16 v4h;
typedef __attribute__((ext_vector_type(8)))  float    v8f;
typedef __attribute__((ext_vector_type(8)))  unsigned short v8s;
typedef __attribute__((ext_vector_type(4)))  unsigned int   u32x4;
typedef __attribute__((ext_vector_type(8)))  int            i32x8;
typedef __attribute__((ext_vector_type(4)))  int            i32x4;

#define WDIM 256
#define CDIM 64
#define QSTR 72    // stride (halves) for [*,64] f16 buffers; 144B rows, 16B aligned
#define ASTR 264   // stride (elems) for [*,256] buffers; 528B rows
#define SCALE 0.125f

__device__ __forceinline__ unsigned short f32_bf16(float f) {
    union { float f; unsigned u; } v; v.f = f;
    unsigned r = (v.u + 0x7FFFu + ((v.u >> 16) & 1u)) >> 16;
    return (unsigned short)r;
}
__device__ __forceinline__ float bf16_f32(unsigned short h) {
    union { unsigned u; float f; } v; v.u = ((unsigned)h) << 16;
    return v.f;
}
__device__ __forceinline__ unsigned lds_off(const void* p) {
    return (unsigned)(unsigned long long)p;   // LDS aperture: addr[31:0] is the LDS offset
}

// TDM: 1-D tensor load, `elems` 4-byte elements, global -> LDS.
__device__ __forceinline__ void tdm_load_1d_f32(unsigned lds_addr, const void* gptr, int elems) {
    unsigned long long ga = (unsigned long long)gptr;
    u32x4 g0;
    g0[0] = 1u;                                   // count=1, user descriptor
    g0[1] = lds_addr;                             // lds_addr [63:32]
    g0[2] = (unsigned)(ga & 0xFFFFFFFFull);       // global_addr lo
    g0[3] = (unsigned)((ga >> 32) & 0x01FFFFFFull) | (2u << 30);  // addr hi | type=2
    i32x8 g1;
    g1[0] = (int)(2u << 16);                      // data_size = 2 (4 bytes)
    g1[1] = (int)((unsigned)(elems & 0xFFFF) << 16);     // tensor_dim0[15:0] @ bits63:48
    g1[2] = (int)(((unsigned)elems >> 16) | (1u << 16)); // tensor_dim0 hi, tensor_dim1 = 1
    g1[3] = (int)((unsigned)(elems & 0xFFFF) << 16);     // tile_dim0 @ bits127:112
    g1[4] = 0;                                    // tile_dim1=0 (unused), tile_dim2=0
    g1[5] = elems;                                // tensor_dim0_stride lo
    g1[6] = 0;
    g1[7] = 0;
    i32x4 z4 = {};
#if __clang_major__ >= 23
    i32x8 z8 = {};
    __builtin_amdgcn_tensor_load_to_lds(g0, g1, z4, z4, z8, 0);
#else
    __builtin_amdgcn_tensor_load_to_lds(g0, g1, z4, z4, 0);
#endif
}

// A fragment (16x32 f16, MxK) from row-major f16 buffer (ISA 7.12.2 wave32 layout).
__device__ __forceinline__ v16h load_a(const _Float16* buf, int stride,
                                       int mBase, int kBase, int lane) {
    int m  = mBase + (lane & 15);
    int k0 = kBase + ((lane >> 4) << 3);
    const _Float16* p = buf + m * stride + k0;
    v8h lo = *(const v8h*)(p);
    v8h hi = *(const v8h*)(p + 16);
    v16h a;
#pragma unroll
    for (int i = 0; i < 8; i++) { a[i] = lo[i]; a[i + 8] = hi[i]; }
    return a;
}

// B fragment (32x16 f16, KxN) from a K-contiguous ([N][K] transposed) buffer.
__device__ __forceinline__ v16h load_b(const _Float16* bufT, int stride,
                                       int nBase, int kBase, int lane) {
    int n  = nBase + (lane & 15);
    int k0 = kBase + ((lane >> 4) << 3);
    const _Float16* p = bufT + n * stride + k0;
    v8h lo = *(const v8h*)(p);
    v8h hi = *(const v8h*)(p + 16);
    v16h b;
#pragma unroll
    for (int i = 0; i < 8; i++) { b[i] = lo[i]; b[i + 8] = hi[i]; }
    return b;
}

struct Stage {
    union {
        struct {
            _Float16 ln[WDIM * QSTR];     // LayerNormed rows (current side)
            _Float16 xs[WDIM * QSTR];     // raw x rows as f16 (current side)
        } s;
        float wstage[4][CDIM * CDIM];     // 64KB f32 weight staging (TDM target)
    } in;
    _Float16 wt[4][CDIM * QSTR];          // W_* transposed [n][k], f16
};

struct Smem {
    union {
        unsigned short attn[WDIM * ASTR]; // 132KB bf16 logits (phase>=2)
        Stage stage;                      // phase 0/1 staging
    } u;
    _Float16 q[2][WDIM * QSTR];           // q_l, q_r row-major [w][c]
    _Float16 vT[2][CDIM * ASTR];          // v_l, v_r transposed [c][w]
    float cstat[8][WDIM];                 // column-stat partial reduction scratch
    float rowMax[WDIM], rowRcp[WDIM];
    float colMax[WDIM], colRcp[WDIM];
    float lnsc[2][CDIM], lnbs[2][CDIM];
    float bq[2][CDIM], bv[2][CDIM];
    float beta[CDIM], gamma[CDIM];
};

__global__ __launch_bounds__(256) void scam_kernel(
    const float* __restrict__ x_l, const float* __restrict__ x_r,
    const float* __restrict__ ln_l_s, const float* __restrict__ ln_l_b,
    const float* __restrict__ ln_r_s, const float* __restrict__ ln_r_b,
    const float* __restrict__ W_ql, const float* __restrict__ b_ql,
    const float* __restrict__ W_qr, const float* __restrict__ b_qr,
    const float* __restrict__ W_vl, const float* __restrict__ b_vl,
    const float* __restrict__ W_vr, const float* __restrict__ b_vr,
    const float* __restrict__ beta, const float* __restrict__ gamma,
    float* __restrict__ out_l, float* __restrict__ out_r)
{
    __shared__ Smem sm;
    const int t    = threadIdx.x;
    const int lane = t & 31;
    const int wave = __builtin_amdgcn_readfirstlane(t >> 5);   // SGPR wave id
    const int bh   = blockIdx.x;                               // 0..1023 = b*H + h
    const size_t tileBase = (size_t)bh * WDIM * CDIM;

    // Prefetch x_r tile row for this thread (consumed in phase 1, side 1).
    __builtin_prefetch(x_r + tileBase + (size_t)t * CDIM, 0, 1);

    // ---------------- Phase 0: TDM weight DMA + params ----------------
    if (wave == 0) {
        const float* Ws[4] = { W_ql, W_qr, W_vl, W_vr };
#pragma unroll
        for (int w = 0; w < 4; w++)
            tdm_load_1d_f32(lds_off(&sm.u.stage.in.wstage[w][0]), Ws[w], CDIM * CDIM);
        __builtin_amdgcn_s_wait_tensorcnt(0);
    }
    if (t < CDIM) {
        sm.lnsc[0][t] = ln_l_s[t]; sm.lnbs[0][t] = ln_l_b[t];
        sm.lnsc[1][t] = ln_r_s[t]; sm.lnbs[1][t] = ln_r_b[t];
        sm.bq[0][t] = b_ql[t]; sm.bq[1][t] = b_qr[t];
        sm.bv[0][t] = b_vl[t]; sm.bv[1][t] = b_vr[t];
        sm.beta[t] = beta[t];  sm.gamma[t] = gamma[t];
    }
    __syncthreads();
    // Convert staged f32 weights -> f16 transposed [n][k]
#pragma unroll
    for (int w = 0; w < 4; w++) {
#pragma unroll
        for (int i = 0; i < 16; i++) {
            int idx = t + 256 * i;             // 0..4095
            int k = idx >> 6, n = idx & 63;    // W is [k][n] row-major
            sm.u.stage.wt[w][n * QSTR + k] = (_Float16)sm.u.stage.in.wstage[w][idx];
        }
    }
    __syncthreads();

    // ---------------- Phase 1: LN + projections (per side) ----------------
    for (int side = 0; side < 2; side++) {
        const float* xp = (side == 0 ? x_l : x_r) + tileBase + (size_t)t * CDIM;
        float4 r[16];
        float s = 0.f;
#pragma unroll
        for (int i = 0; i < 16; i++) {
            r[i] = ((const float4*)xp)[i];
            s += r[i].x + r[i].y + r[i].z + r[i].w;
        }
        float mean = s * (1.f / 64.f);
        float var = 0.f;
#pragma unroll
        for (int i = 0; i < 16; i++) {
            float dx = r[i].x - mean, dy = r[i].y - mean;
            float dz = r[i].z - mean, dw = r[i].w - mean;
            var += dx * dx + dy * dy + dz * dz + dw * dw;
        }
        float rstd = rsqrtf(var * (1.f / 64.f) + 1e-6f);
#pragma unroll
        for (int i = 0; i < 16; i++) {
            int c = i * 4;
            v4h xv, lv;
            xv[0] = (_Float16)r[i].x; xv[1] = (_Float16)r[i].y;
            xv[2] = (_Float16)r[i].z; xv[3] = (_Float16)r[i].w;
            lv[0] = (_Float16)((r[i].x - mean) * rstd * sm.lnsc[side][c + 0] + sm.lnbs[side][c + 0]);
            lv[1] = (_Float16)((r[i].y - mean) * rstd * sm.lnsc[side][c + 1] + sm.lnbs[side][c + 1]);
            lv[2] = (_Float16)((r[i].z - mean) * rstd * sm.lnsc[side][c + 2] + sm.lnbs[side][c + 2]);
            lv[3] = (_Float16)((r[i].w - mean) * rstd * sm.lnsc[side][c + 3] + sm.lnbs[side][c + 3]);
            *(v4h*)&sm.u.stage.in.s.xs[t * QSTR + c] = xv;
            *(v4h*)&sm.u.stage.in.s.ln[t * QSTR + c] = lv;
        }
        __syncthreads();

        // q = ln @ W_q + b_q (waves 0-3) ; v = xs @ W_v + b_v (waves 4-7).
        // A fragment hoisted: loaded once per mTile, reused across the 4 nTiles.
        if (wave < 4) {
            const _Float16* A = sm.u.stage.in.s.ln;
            const _Float16* B = sm.u.stage.wt[side];
            const float* bias = sm.bq[side];
#pragma unroll
            for (int mSub = 0; mSub < 4; mSub++) {
                int mT = wave * 4 + mSub;
                v16h a0 = load_a(A, QSTR, mT * 16, 0,  lane);
                v16h a1 = load_a(A, QSTR, mT * 16, 32, lane);
#pragma unroll
                for (int nT = 0; nT < 4; nT++) {
                    v16h b0 = load_b(B, QSTR, nT * 16, 0,  lane);
                    v16h b1 = load_b(B, QSTR, nT * 16, 32, lane);
                    v8f acc = {};
                    acc = __builtin_amdgcn_wmma_f32_16x16x32_f16(false, a0, false, b0, (short)0, acc, false, false);
                    acc = __builtin_amdgcn_wmma_f32_16x16x32_f16(false, a1, false, b1, (short)0, acc, false, false);
                    int n    = nT * 16 + (lane & 15);
                    int mOff = mT * 16 + ((lane >> 4) << 3);
                    float bs = bias[n];
#pragma unroll
                    for (int rI = 0; rI < 8; rI++)
                        sm.q[side][(mOff + rI) * QSTR + n] = (_Float16)(acc[rI] + bs);
                }
            }
        } else {
            const _Float16* A = sm.u.stage.in.s.xs;
            const _Float16* B = sm.u.stage.wt[2 + side];
            const float* bias = sm.bv[side];
#pragma unroll
            for (int mSub = 0; mSub < 4; mSub++) {
                int mT = (wave - 4) * 4 + mSub;
                v16h a0 = load_a(A, QSTR, mT * 16, 0,  lane);
                v16h a1 = load_a(A, QSTR, mT * 16, 32, lane);
#pragma unroll
                for (int nT = 0; nT < 4; nT++) {
                    v16h b0 = load_b(B, QSTR, nT * 16, 0,  lane);
                    v16h b1 = load_b(B, QSTR, nT * 16, 32, lane);
                    v8f acc = {};
                    acc = __builtin_amdgcn_wmma_f32_16x16x32_f16(false, a0, false, b0, (short)0, acc, false, false);
                    acc = __builtin_amdgcn_wmma_f32_16x16x32_f16(false, a1, false, b1, (short)0, acc, false, false);
                    int n    = nT * 16 + (lane & 15);
                    int mOff = mT * 16 + ((lane >> 4) << 3);
                    float bs = bias[n];
                    v8h pk;
#pragma unroll
                    for (int rI = 0; rI < 8; rI++) pk[rI] = (_Float16)(acc[rI] + bs);
                    *(v8h*)&sm.vT[side][n * ASTR + mOff] = pk;   // contiguous b128 store
                }
            }
        }
        __syncthreads();
    }

    // ---------------- Phase 2: attn = q_l @ q_r^T * SCALE (bf16) ----------------
#pragma unroll
    for (int mi = 0; mi < 2; mi++) {
        int mT = wave * 2 + mi;
        v16h a0 = load_a(sm.q[0], QSTR, mT * 16, 0,  lane);
        v16h a1 = load_a(sm.q[0], QSTR, mT * 16, 32, lane);
        for (int nT = 0; nT < 16; nT++) {
            v16h b0 = load_b(sm.q[1], QSTR, nT * 16, 0,  lane);
            v16h b1 = load_b(sm.q[1], QSTR, nT * 16, 32, lane);
            v8f acc = {};
            acc = __builtin_amdgcn_wmma_f32_16x16x32_f16(false, a0, false, b0, (short)0, acc, false, false);
            acc = __builtin_amdgcn_wmma_f32_16x16x32_f16(false, a1, false, b1, (short)0, acc, false, false);
            int n    = nT * 16 + (lane & 15);
            int mOff = mT * 16 + ((lane >> 4) << 3);
#pragma unroll
            for (int rI = 0; rI < 8; rI++)
                sm.u.attn[(mOff + rI) * ASTR + n] = f32_bf16(acc[rI] * SCALE);
        }
    }
    __syncthreads();

    // ---------------- Phase 3: softmax stats ----------------
    {   // Row stats: thread t scans row t with b128 chunks (rotated start).
        const v8s* rowp = (const v8s*)&sm.u.attn[t * ASTR];
        int rot = (2 * t) & 31;
        float mx = -1e30f;
        for (int i = 0; i < 32; i++) {
            v8s ch = rowp[(i + rot) & 31];
#pragma unroll
            for (int j = 0; j < 8; j++) mx = fmaxf(mx, bf16_f32(ch[j]));
        }
        float sum = 0.f;
        for (int i = 0; i < 32; i++) {
            v8s ch = rowp[(i + rot) & 31];
#pragma unroll
            for (int j = 0; j < 8; j++) sum += __expf(bf16_f32(ch[j]) - mx);
        }
        sm.rowMax[t] = mx; sm.rowRcp[t] = 1.f / sum;
    }
    {   // Column stats: grouped. Thread owns cols [cb*8, cb*8+8), rows [rg*32, rg*32+32).
        int cb = lane, rg = wave;
        int c0 = cb * 8;
        float m8[8];
#pragma unroll
        for (int j = 0; j < 8; j++) m8[j] = -1e30f;
        for (int w = rg * 32; w < rg * 32 + 32; w++) {
            v8s ch = *(const v8s*)&sm.u.attn[w * ASTR + c0];
#pragma unroll
            for (int j = 0; j < 8; j++) m8[j] = fmaxf(m8[j], bf16_f32(ch[j]));
        }
#pragma unroll
        for (int j = 0; j < 8; j++) sm.cstat[rg][c0 + j] = m8[j];
        __syncthreads();
        float mc = sm.cstat[0][t];
#pragma unroll
        for (int g = 1; g < 8; g++) mc = fmaxf(mc, sm.cstat[g][t]);
        sm.colMax[t] = mc;
        __syncthreads();
        float mx8[8], s8[8];
#pragma unroll
        for (int j = 0; j < 8; j++) { mx8[j] = sm.colMax[c0 + j]; s8[j] = 0.f; }
        for (int w = rg * 32; w < rg * 32 + 32; w++) {
            v8s ch = *(const v8s*)&sm.u.attn[w * ASTR + c0];
#pragma unroll
            for (int j = 0; j < 8; j++) s8[j] += __expf(bf16_f32(ch[j]) - mx8[j]);
        }
#pragma unroll
        for (int j = 0; j < 8; j++) sm.cstat[rg][c0 + j] = s8[j];
        __syncthreads();
        float sc = sm.cstat[0][t];
#pragma unroll
        for (int g = 1; g < 8; g++) sc += sm.cstat[g][t];
        sm.colRcp[t] = 1.f / sc;
    }
    __syncthreads();

    // ---------------- Phase 4: out_l = x_l + beta * softmax_row(attn) @ v_r ----------------
#pragma unroll
    for (int mi = 0; mi < 2; mi++) {
        int mT = wave * 2 + mi;
        v8f acc[4] = {};
        int m = mT * 16 + (lane & 15);
        float mx = sm.rowMax[m], rs = sm.rowRcp[m];
        for (int k = 0; k < 8; k++) {
            int k0 = k * 32 + ((lane >> 4) << 3);
            const v8s* pv = (const v8s*)&sm.u.attn[m * ASTR + k0];
            v8s lo = pv[0], hi = pv[2];
            v16h a;
#pragma unroll
            for (int j = 0; j < 8; j++) {
                a[j]     = (_Float16)(__expf(bf16_f32(lo[j]) - mx) * rs);
                a[j + 8] = (_Float16)(__expf(bf16_f32(hi[j]) - mx) * rs);
            }
#pragma unroll
            for (int nT = 0; nT < 4; nT++) {
                v16h b = load_b(sm.vT[1], ASTR, nT * 16, k * 32, lane);
                acc[nT] = __builtin_amdgcn_wmma_f32_16x16x32_f16(
                    false, a, false, b, (short)0, acc[nT], false, false);
            }
        }
        int mOff = mT * 16 + ((lane >> 4) << 3);
#pragma unroll
        for (int nT = 0; nT < 4; nT++) {
            int c = nT * 16 + (lane & 15);
            float bta = sm.beta[c];
#pragma unroll
            for (int rI = 0; rI < 8; rI++) {
                size_t idx = tileBase + (size_t)(mOff + rI) * CDIM + c;
                out_l[idx] = x_l[idx] + bta * acc[nT][rI];
            }
        }
    }

    // ---------------- Phase 5: out_r = x_r + gamma * softmax_col(attn)^T @ v_l ----------------
#pragma unroll
    for (int mi = 0; mi < 2; mi++) {
        int mT = wave * 2 + mi;
        v8f acc[4] = {};
        int n = mT * 16 + (lane & 15);       // output row = W_r index
        float mx = sm.colMax[n], rs = sm.colRcp[n];
        for (int k = 0; k < 8; k++) {
            int k0 = k * 32 + ((lane >> 4) << 3);
            v16h a;
#pragma unroll
            for (int j = 0; j < 8; j++) {
                a[j]     = (_Float16)(__expf(bf16_f32(sm.u.attn[(k0 + j)      * ASTR + n]) - mx) * rs);
                a[j + 8] = (_Float16)(__expf(bf16_f32(sm.u.attn[(k0 + 16 + j) * ASTR + n]) - mx) * rs);
            }
#pragma unroll
            for (int nT = 0; nT < 4; nT++) {
                v16h b = load_b(sm.vT[0], ASTR, nT * 16, k * 32, lane);
                acc[nT] = __builtin_amdgcn_wmma_f32_16x16x32_f16(
                    false, a, false, b, (short)0, acc[nT], false, false);
            }
        }
        int mOff = mT * 16 + ((lane >> 4) << 3);
#pragma unroll
        for (int nT = 0; nT < 4; nT++) {
            int c = nT * 16 + (lane & 15);
            float gma = sm.gamma[c];
#pragma unroll
            for (int rI = 0; rI < 8; rI++) {
                size_t idx = tileBase + (size_t)(mOff + rI) * CDIM + c;
                out_r[idx] = x_r[idx] + gma * acc[nT][rI];
            }
        }
    }
}

extern "C" void kernel_launch(void* const* d_in, const int* in_sizes, int n_in,
                              void* d_out, int out_size, void* d_ws, size_t ws_size,
                              hipStream_t stream) {
    (void)in_sizes; (void)n_in; (void)out_size; (void)d_ws; (void)ws_size;
    const float* x_l    = (const float*)d_in[0];
    const float* x_r    = (const float*)d_in[1];
    const float* ln_l_s = (const float*)d_in[2];
    const float* ln_l_b = (const float*)d_in[3];
    const float* ln_r_s = (const float*)d_in[4];
    const float* ln_r_b = (const float*)d_in[5];
    const float* W_ql   = (const float*)d_in[6];
    const float* b_ql   = (const float*)d_in[7];
    const float* W_qr   = (const float*)d_in[8];
    const float* b_qr   = (const float*)d_in[9];
    const float* W_vl   = (const float*)d_in[10];
    const float* b_vl   = (const float*)d_in[11];
    const float* W_vr   = (const float*)d_in[12];
    const float* b_vr   = (const float*)d_in[13];
    const float* beta   = (const float*)d_in[14];
    const float* gamma  = (const float*)d_in[15];

    float* out_l = (float*)d_out;
    float* out_r = out_l + (size_t)4 * 256 * 256 * 64;

    scam_kernel<<<1024, 256, 0, stream>>>(
        x_l, x_r, ln_l_s, ln_l_b, ln_r_s, ln_r_b,
        W_ql, b_ql, W_qr, b_qr, W_vl, b_vl, W_vr, b_vr,
        beta, gamma, out_l, out_r);
}
